// GroupedQueryAttention_37349035606742
// MI455X (gfx1250) — matrix operand order repaired and checked by reading, not compiled, not run on hardware
//
#include <hip/hip_runtime.h>
#include <hip/hip_bf16.h>

// Problem constants (match reference)
#define S_LEN 2048
#define DM    1024
#define NH    16
#define NKV   4
#define DK    64
#define BB    2

typedef __attribute__((ext_vector_type(16))) __bf16 v16bf;
typedef __attribute__((ext_vector_type(8)))  float  v8f;

// ---- helpers -------------------------------------------------------------

static __device__ __forceinline__ __bf16 f2bf(float f) {
  unsigned u = __builtin_bit_cast(unsigned, f);
  unsigned r = u + 0x7FFFu + ((u >> 16) & 1u);   // round-to-nearest-even
  unsigned short h = (unsigned short)(r >> 16);
  return __builtin_bit_cast(__bf16, h);
}
static __device__ __forceinline__ __bf16 to_bf(float f)  { return f2bf(f); }
static __device__ __forceinline__ __bf16 to_bf(__bf16 x) { return x; }

// wave32 cross-lane xor exchange via ds_bpermute (no warpSize assumptions)
static __device__ __forceinline__ float lane_xor_f(float x, int mask) {
  int lane = (int)(threadIdx.x & 31);
  int v = __builtin_amdgcn_ds_bpermute(((lane ^ mask) << 2),
                                       __builtin_bit_cast(int, x));
  return __builtin_bit_cast(float, v);
}
// reduce within 16-lane halves (masks 1,2,4,8 stay inside a half)
static __device__ __forceinline__ float half_max(float x) {
#pragma unroll
  for (int m = 1; m <= 8; m <<= 1) x = fmaxf(x, lane_xor_f(x, m));
  return x;
}
static __device__ __forceinline__ float half_sum(float x) {
#pragma unroll
  for (int m = 1; m <= 8; m <<= 1) x += lane_xor_f(x, m);
  return x;
}

// CDNA5 async global->LDS copy, 16 bytes per lane (VFLAT async encoding:
// vdst = per-lane LDS byte address, vaddr = per-lane 64-bit global address).
static __device__ __forceinline__ void async_copy_b128(unsigned lds_addr,
                                                       const void* gaddr) {
  asm volatile("global_load_async_to_lds_b128 %0, %1, off"
               :: "v"(lds_addr), "v"((unsigned long long)(size_t)gaddr)
               : "memory");
}

// ---- weight transpose+convert: Wt[n][k] = bf16(W[k][n]) ------------------
// One-time pass; makes all GEMM B-fragment loads contiguous 16xbf16 (b128).
__global__ void __launch_bounds__(256)
wt_transpose_bf16(const float* __restrict__ W, __bf16* __restrict__ Wt,
                  int Kdim, int N) {
  int idx = blockIdx.x * 256 + threadIdx.x;     // k fastest -> coalesced store
  if (idx >= Kdim * N) return;
  int n = idx / Kdim, k = idx - n * Kdim;
  Wt[idx] = f2bf(W[(size_t)k * N + n]);
}

// ---- GEMM: C[32 x 64 per wave] = A @ Wt^T + bias (bf16 WMMA, f32 accum) --
// 128-thread blocks: 4 waves, each wave a 32x64 tile of a 32x256 block tile.
// Wt is pre-transposed bf16 [N][K]; A is fp32 (activations) or bf16 (ctx).
// mode 0: bf16 out row-major [M][N] (scaled by bscale) and/or f32 row-major
// mode 1: f32+bf16 out remapped to KV cache layout [B,KVH,S,DK]
// mode 2: f32 out [B,KVH,S,DK], bf16 out transposed [B,KVH,DK,S] (for V)
template <typename AT>
__global__ void __launch_bounds__(128)
gemm_bias_wmma(const AT* __restrict__ A, const __bf16* __restrict__ Wt,
               const float* __restrict__ bias, int Kdim, int N,
               __bf16* __restrict__ outb, float* __restrict__ outf,
               int mode, float bscale) {
  const int lane = threadIdx.x & 31;
  const int wid  = threadIdx.x >> 5;
  const int half = lane >> 4, lm = lane & 15;
  const int row0 = blockIdx.y * 32;
  const int col0 = blockIdx.x * 256 + wid * 64;

  v8f acc[2][4];
#pragma unroll
  for (int rt = 0; rt < 2; ++rt)
#pragma unroll
    for (int t = 0; t < 4; ++t)
#pragma unroll
      for (int r = 0; r < 8; ++r) acc[rt][t][r] = 0.0f;

  const AT* Ar[2] = { A + (size_t)(row0 + lm) * Kdim,
                      A + (size_t)(row0 + 16 + lm) * Kdim };

  for (int k0 = 0; k0 < Kdim; k0 += 32) {
    if (k0 + 32 < Kdim) {                         // cover HBM latency
      __builtin_prefetch(Ar[0] + k0 + 32, 0, 1);
      __builtin_prefetch(Ar[1] + k0 + 32, 0, 1);
    }
    // A fragments: m = lm, k = half*16 + i  (16-elem runs)
    v16bf a[2];
#pragma unroll
    for (int rt = 0; rt < 2; ++rt)
#pragma unroll
      for (int i = 0; i < 16; ++i) {
        int kk = (half << 4) + i;
        a[rt][i] = to_bf(Ar[rt][k0 + kk]);
      }
#pragma unroll
    for (int t = 0; t < 4; ++t) {
      // B fragment from Wt[n][k]: n = lm (row of Wt), k = half*16 + i
      // -> 16 contiguous bf16 per lane (one b128 pair)
      const __bf16* wrow = Wt + (size_t)(col0 + (t << 4) + lm) * Kdim + k0 +
                           (half << 4);
      v16bf bw;
#pragma unroll
      for (int i = 0; i < 16; ++i) bw[i] = wrow[i];
      acc[0][t] = __builtin_amdgcn_wmma_f32_16x16x32_bf16(
          false, a[0], false, bw, (short)0, acc[0][t], false, false);
      acc[1][t] = __builtin_amdgcn_wmma_f32_16x16x32_bf16(
          false, a[1], false, bw, (short)0, acc[1][t], false, false);
    }
  }

#pragma unroll
  for (int rt = 0; rt < 2; ++rt)
#pragma unroll
    for (int t = 0; t < 4; ++t) {
      int col = col0 + (t << 4) + lm;
      float bvv = bias[col];
#pragma unroll
      for (int r = 0; r < 8; ++r) {
        float v = acc[rt][t][r] + bvv;
        int row = row0 + (rt << 4) + r + (half << 3);
        if (mode == 0) {
          size_t idx = (size_t)row * N + col;
          if (outb) outb[idx] = f2bf(v * bscale);
          if (outf) outf[idx] = v;
        } else {
          int bb = row / S_LEN, srow = row - bb * S_LEN;
          int kvh = col >> 6, d = col & 63;
          size_t idx = (((size_t)(bb * NKV + kvh) * S_LEN + srow) * DK + d);
          outf[idx] = v;                                   // fp32 cache out
          if (mode == 1)
            outb[idx] = f2bf(v);                           // K: [B,KVH,S,DK]
          else
            outb[((size_t)(bb * NKV + kvh) * DK + d) * S_LEN + srow] =
                f2bf(v);                                   // V: [B,KVH,DK,S]
        }
      }
    }
}

// ---- Flash GQA: one wave per (b, head, 32-query tile) --------------------
// K/V chunk tiles are staged global->LDS with CDNA5 async copies, double
// buffered on ASYNCcnt so chunk i+1 streams in while chunk i runs on WMMA.
__global__ void __launch_bounds__(32)
gqa_flash_wmma(const __bf16* __restrict__ Qb,   // [B,S,DM], pre-scaled 1/8
               const __bf16* __restrict__ Kb,   // [B,KVH,S,DK]
               const __bf16* __restrict__ Vt,   // [B,KVH,DK,S]
               __bf16* __restrict__ ctx) {      // [B,S,DM]
  __shared__ __bf16 Ktile[2][32 * DK];          // [buf][key][d]   4 KB each
  __shared__ __bf16 Vtile[2][DK * 32];          // [buf][d][key]   4 KB each
  __shared__ __bf16 Plds[32 * 32];              // P transpose staging (2 KB)
  const int lane = threadIdx.x;
  const int half = lane >> 4, lm = lane & 15;
  const int q0 = blockIdx.x * 32;
  const int h  = blockIdx.y;
  const int b  = blockIdx.z;
  const int kvh = h >> 2;                       // G = 4
  const size_t kbase = (size_t)(b * NKV + kvh) * S_LEN * DK;
  const size_t vbase = (size_t)(b * NKV + kvh) * DK * S_LEN;

  const char* Kg = (const char*)(Kb + kbase);   // chunk rows are contiguous
  const char* Vg = (const char*)(Vt + vbase);   // 64 rows, stride S_LEN*2 B
  const unsigned ldsK[2] = { (unsigned)(size_t)(void*)&Ktile[0][0],
                             (unsigned)(size_t)(void*)&Ktile[1][0] };
  const unsigned ldsV[2] = { (unsigned)(size_t)(void*)&Vtile[0][0],
                             (unsigned)(size_t)(void*)&Vtile[1][0] };

  // issue one chunk's K+V tiles: 16 async b128 ops (8 K + 8 V) per lane set
  auto issue_chunk = [&](int key0, int buf) {
    const char* gk = Kg + (size_t)key0 * DK * 2;       // 4 KB contiguous
#pragma unroll
    for (int j = 0; j < 8; ++j) {
      int seg = j * 32 + lane;                         // 256 x 16B segments
      async_copy_b128(ldsK[buf] + seg * 16, gk + seg * 16);
    }
#pragma unroll
    for (int j = 0; j < 8; ++j) {
      int seg = j * 32 + lane;                         // seg = d*4 + part
      int d = seg >> 2, part = seg & 3;
      async_copy_b128(ldsV[buf] + d * 64 + part * 16,
                      Vg + ((size_t)d * S_LEN + key0) * 2 + part * 16);
    }
  };

  // Q A-fragments: 2 row-tiles x 2 dk-chunks, resident for whole KV sweep
  v16bf aq[2][2];
#pragma unroll
  for (int rt = 0; rt < 2; ++rt)
#pragma unroll
    for (int c = 0; c < 2; ++c)
#pragma unroll
      for (int i = 0; i < 16; ++i) {
        int d = (c << 5) + (half << 4) + i;
        int row = q0 + (rt << 4) + lm;
        aq[rt][c][i] = Qb[((size_t)b * S_LEN + row) * DM + h * DK + d];
      }

  float mrow[2][8], lrow[2][8];
  v8f acc[2][4];
#pragma unroll
  for (int rt = 0; rt < 2; ++rt) {
#pragma unroll
    for (int r = 0; r < 8; ++r) { mrow[rt][r] = -3.0e38f; lrow[rt][r] = 0.0f; }
#pragma unroll
    for (int t = 0; t < 4; ++t)
#pragma unroll
      for (int r = 0; r < 8; ++r) acc[rt][t][r] = 0.0f;
  }

  issue_chunk(0, 0);                             // prologue: chunk 0 in buf 0

  for (int key0 = 0; key0 < S_LEN; key0 += 32) {
    const int buf = (key0 >> 5) & 1;
    if (key0 + 32 < S_LEN) {
      issue_chunk(key0 + 32, buf ^ 1);           // stream next chunk
      // async loads complete in order: <=16 outstanding => chunk resident
      asm volatile("s_wait_asynccnt 0x10" ::: "memory");
    } else {
      asm volatile("s_wait_asynccnt 0x0" ::: "memory");
    }

    // ---- scores: S[32q x 32k], each K fragment feeds both row-tiles ----
    v8f s[2][2];
#pragma unroll
    for (int rt = 0; rt < 2; ++rt)
#pragma unroll
      for (int t = 0; t < 2; ++t)
#pragma unroll
        for (int r = 0; r < 8; ++r) s[rt][t][r] = 0.0f;
#pragma unroll
    for (int t = 0; t < 2; ++t)
#pragma unroll
      for (int c = 0; c < 2; ++c) {
        const __bf16* krow =                      // 16 contiguous bf16/lane
            &Ktile[buf][((t << 4) + lm) * DK + (c << 5) + (half << 4)];
        v16bf kb;
#pragma unroll
        for (int i = 0; i < 16; ++i) kb[i] = krow[i];
        s[0][t] = __builtin_amdgcn_wmma_f32_16x16x32_bf16(
            false, aq[0][c], false, kb, (short)0, s[0][t], false, false);
        s[1][t] = __builtin_amdgcn_wmma_f32_16x16x32_bf16(
            false, aq[1][c], false, kb, (short)0, s[1][t], false, false);
      }

    // ---- online softmax (scale already folded into Qb) ----
#pragma unroll
    for (int rt = 0; rt < 2; ++rt)
#pragma unroll
      for (int r = 0; r < 8; ++r) {
        float s0 = s[rt][0][r], s1 = s[rt][1][r];
        float mx = half_max(fmaxf(s0, s1));
        float mn = fmaxf(mrow[rt][r], mx);
        float alpha = __expf(mrow[rt][r] - mn);
        float e0 = __expf(s0 - mn), e1 = __expf(s1 - mn);
        lrow[rt][r] = lrow[rt][r] * alpha + half_sum(e0 + e1);
        mrow[rt][r] = mn;
#pragma unroll
        for (int t = 0; t < 4; ++t) acc[rt][t][r] *= alpha;
        int m = (rt << 4) + r + (half << 3);      // C-layout row
        Plds[m * 32 + lm]      = f2bf(e0);
        Plds[m * 32 + 16 + lm] = f2bf(e1);
      }
    asm volatile("s_wait_dscnt 0" ::: "memory");  // P fully in LDS

    // ---- transpose P into A-fragment layout ----
    v16bf pa[2];
#pragma unroll
    for (int rt = 0; rt < 2; ++rt)
#pragma unroll
      for (int i = 0; i < 16; ++i) {
        int k = (half << 4) + i;
        pa[rt][i] = Plds[((rt << 4) + lm) * 32 + k];
      }

    // ---- ctx += P(32x32) . V(32x64); each V fragment feeds both tiles ----
#pragma unroll
    for (int n0 = 0; n0 < 4; ++n0) {
      const __bf16* vrow =                        // 16 contiguous bf16/lane
          &Vtile[buf][((n0 << 4) + lm) * 32 + (half << 4)];
      v16bf vb;
#pragma unroll
      for (int i = 0; i < 16; ++i) vb[i] = vrow[i];
      acc[0][n0] = __builtin_amdgcn_wmma_f32_16x16x32_bf16(
          false, pa[0], false, vb, (short)0, acc[0][n0], false, false);
      acc[1][n0] = __builtin_amdgcn_wmma_f32_16x16x32_bf16(
          false, pa[1], false, vb, (short)0, acc[1][n0], false, false);
    }
  }

  // ---- normalize and emit ctx (bf16) for the output projection ----
#pragma unroll
  for (int rt = 0; rt < 2; ++rt)
#pragma unroll
    for (int n0 = 0; n0 < 4; ++n0)
#pragma unroll
      for (int r = 0; r < 8; ++r) {
        float o = acc[rt][n0][r] / lrow[rt][r];
        int srow = q0 + (rt << 4) + r + (half << 3);
        ctx[((size_t)b * S_LEN + srow) * DM + h * DK + (n0 << 4) + lm] =
            f2bf(o);
      }
}

// ---- launcher ------------------------------------------------------------
extern "C" void kernel_launch(void* const* d_in, const int* in_sizes, int n_in,
                              void* d_out, int out_size, void* d_ws, size_t ws_size,
                              hipStream_t stream) {
  (void)in_sizes; (void)n_in; (void)out_size; (void)ws_size;
  const float* query    = (const float*)d_in[0];
  const float* key_in   = (const float*)d_in[1];
  const float* value_in = (const float*)d_in[2];
  const float* Wq = (const float*)d_in[3];
  const float* bq = (const float*)d_in[4];
  const float* Wk = (const float*)d_in[5];
  const float* bk = (const float*)d_in[6];
  const float* Wv = (const float*)d_in[7];
  const float* bv = (const float*)d_in[8];
  const float* Wo = (const float*)d_in[9];
  const float* bo = (const float*)d_in[10];

  float* out  = (float*)d_out;                               // [B,S,DM]
  float* outK = out  + (size_t)BB * S_LEN * DM;              // [B,KVH,S,DK]
  float* outV = outK + (size_t)BB * NKV * S_LEN * DK;        // [B,KVH,S,DK]

  char* ws = (char*)d_ws;
  __bf16* Qb  = (__bf16*)ws; ws += (size_t)BB * S_LEN * DM * 2;        // 8 MB
  __bf16* Kb  = (__bf16*)ws; ws += (size_t)BB * NKV * S_LEN * DK * 2;  // 2 MB
  __bf16* Vt  = (__bf16*)ws; ws += (size_t)BB * NKV * S_LEN * DK * 2;  // 2 MB
  __bf16* Cb  = (__bf16*)ws; ws += (size_t)BB * S_LEN * DM * 2;        // 8 MB
  __bf16* Wqt = (__bf16*)ws; ws += (size_t)DM * DM * 2;                // 2 MB
  __bf16* Wkt = (__bf16*)ws; ws += (size_t)DM * NKV * DK * 2;          // .5 MB
  __bf16* Wvt = (__bf16*)ws; ws += (size_t)DM * NKV * DK * 2;          // .5 MB
  __bf16* Wot = (__bf16*)ws;                                           // 2 MB

  const int Mrows = BB * S_LEN;                              // 4096
  dim3 gblk(128, 1, 1);

  // One-time weight transpose+convert (bf16, [N][K])
  wt_transpose_bf16<<<(DM * DM + 255) / 256, 256, 0, stream>>>(Wq, Wqt, DM, DM);
  wt_transpose_bf16<<<(DM * NKV * DK + 255) / 256, 256, 0, stream>>>(Wk, Wkt, DM, NKV * DK);
  wt_transpose_bf16<<<(DM * NKV * DK + 255) / 256, 256, 0, stream>>>(Wv, Wvt, DM, NKV * DK);
  wt_transpose_bf16<<<(DM * DM + 255) / 256, 256, 0, stream>>>(Wo, Wot, DM, DM);

  // Q/K/V projections (K,V also write fp32 cache outputs; Q folds 1/sqrt(dk))
  gemm_bias_wmma<float><<<dim3(DM / 256, Mrows / 32), gblk, 0, stream>>>(
      query, Wqt, bq, DM, DM, Qb, nullptr, 0, 0.125f);
  gemm_bias_wmma<float><<<dim3(1, Mrows / 32), gblk, 0, stream>>>(
      key_in, Wkt, bk, DM, NKV * DK, Kb, outK, 1, 1.0f);
  gemm_bias_wmma<float><<<dim3(1, Mrows / 32), gblk, 0, stream>>>(
      value_in, Wvt, bv, DM, NKV * DK, Vt, outV, 2, 1.0f);

  // Flash grouped-query attention (32 queries per wave, async K/V staging)
  gqa_flash_wmma<<<dim3(S_LEN / 32, NH, BB), dim3(32, 1, 1), 0, stream>>>(
      Qb, Kb, Vt, Cb);

  // Output projection
  gemm_bias_wmma<__bf16><<<dim3(DM / 256, Mrows / 32), gblk, 0, stream>>>(
      Cb, Wot, bo, DM, DM, nullptr, out, 0, 1.0f);
}
